// KNN_68204080660530
// MI455X (gfx1250) — matrix operand order repaired and checked by reading, not compiled, not run on hardware
//
#include <hip/hip_runtime.h>
#include <hip/hip_bf16.h>

// Problem constants (from the reference)
#define NN 12288   // nodes
#define DD 128     // embedding dim
#define KK 32      // top-K per row

// GEMM B-staging geometry
#define CHUNK_TILES 4                        // column tiles per LDS chunk
#define CHUNK_ROWS  (CHUNK_TILES * 16)       // 64 xn rows per chunk
#define CHUNK_BYTES (CHUNK_ROWS * DD * 2)    // 16 KB per buffer
#define NCHUNK      (NN / CHUNK_ROWS)        // 192 chunks

typedef __attribute__((ext_vector_type(16))) __bf16 v16bf;
typedef __attribute__((ext_vector_type(8)))  float  v8f;

union Frag16 {
    v16bf v;
    uint4 q[2];
};

// ---------------------------------------------------------------------------
// Kernel 1: row L2-normalize x (f32) -> xn (bf16, RNE) in workspace
// One 128-thread block (4 wave32) per row.
// ---------------------------------------------------------------------------
__global__ __launch_bounds__(128)
void knn_normalize_kernel(const float* __restrict__ x,
                          unsigned short* __restrict__ xn) {
    const int row = blockIdx.x;
    const int t   = threadIdx.x;

    float v = x[(size_t)row * DD + t];
    float s = v * v;

    #pragma unroll
    for (int off = 16; off >= 1; off >>= 1)
        s += __shfl_xor(s, off, 32);

    __shared__ float wsum[4];
    if ((t & 31) == 0) wsum[t >> 5] = s;
    __syncthreads();

    const float tot = wsum[0] + wsum[1] + wsum[2] + wsum[3];
    const float nrm = fmaxf(sqrtf(tot), 1e-12f);
    const float r   = v / nrm;

    // f32 -> bf16 round-to-nearest-even
    unsigned u   = __float_as_uint(r);
    unsigned rnd = u + 0x7FFFu + ((u >> 16) & 1u);
    xn[(size_t)row * DD + t] = (unsigned short)(rnd >> 16);
}

// ---------------------------------------------------------------------------
// CDNA5 async copy: global -> LDS, 16 bytes per lane, tracked by ASYNCcnt.
// lds_off is the raw LDS byte offset (dynamic-LDS base == groupstaticsize).
// ---------------------------------------------------------------------------
__device__ __forceinline__ void async_copy_b128(unsigned lds_off, const void* g) {
    asm volatile("global_load_async_to_lds_b128 %0, %1, off"
                 :: "v"(lds_off), "v"(g) : "memory");
}

__device__ __forceinline__ void wait_async0() {
    asm volatile("s_wait_asynccnt 0x0" ::: "memory");
}

// ---------------------------------------------------------------------------
// Fragment loaders per CDNA5 ISA 16-bit WMMA VGPR layouts (wave32).
// A (16x32, MxK): lane l -> row M=l%16; K-runs of 8 at s and s+16,
//                 s = (l&16) ? 8 : 0.
// B (32x16, KxN): lane l -> col N=l%16; K-runs of 8 at s and s+16,
//                 s = (l&16) ? 8 : 0.
// All offsets are multiples of 8 bf16 = 16 bytes -> aligned b128 loads.
// ---------------------------------------------------------------------------
__device__ __forceinline__ v16bf load_a_frag(const unsigned short* rowp,
                                             int kbase, int lane) {
    Frag16 f;
    const int s = (lane & 16) ? 8 : 0;
    f.q[0] = *(const uint4*)(rowp + kbase + s);
    f.q[1] = *(const uint4*)(rowp + kbase + s + 16);
    return f.v;
}

__device__ __forceinline__ v16bf load_b_frag(const unsigned short* rowp,
                                             int kbase, int lane) {
    Frag16 f;
    const int s = (lane & 16) ? 8 : 0;
    f.q[0] = *(const uint4*)(rowp + kbase + s);
    f.q[1] = *(const uint4*)(rowp + kbase + s + 16);
    return f.v;
}

// ---------------------------------------------------------------------------
// Kernel 2: dist = xn @ xn^T via v_wmma_f32_16x16x32_bf16.
// 8 waves/block, each wave owns a 16-row stripe; the block sweeps all column
// tiles together. B tiles are staged once per block into a double-buffered
// LDS window via global_load_async_to_lds_b128 (8x less L2 traffic than
// per-wave global loads). A-fragments (full K=128) stay resident in VGPRs.
// ---------------------------------------------------------------------------
__global__ __launch_bounds__(256)
void knn_gemm_kernel(const unsigned short* __restrict__ xn,
                     float* __restrict__ out) {
    extern __shared__ unsigned short smem[];   // 2 x CHUNK_BYTES B staging
    const unsigned lds_base = __builtin_amdgcn_groupstaticsize();

    const int t    = threadIdx.x;
    const int lane = t & 31;
    const int wave = t >> 5;
    const int i0   = (blockIdx.x * 8 + wave) * 16;

    // resident A fragments for this wave's 16 rows (K = 0..127)
    const unsigned short* pa = xn + (size_t)(i0 + (lane & 15)) * DD;
    const v16bf a0 = load_a_frag(pa,  0, lane);
    const v16bf a1 = load_a_frag(pa, 32, lane);
    const v16bf a2 = load_a_frag(pa, 64, lane);
    const v16bf a3 = load_a_frag(pa, 96, lane);

    const int cn = lane & 15;        // column within tile
    const int mh = (lane >> 4) << 3; // C-layout row offset (0 or 8)

    // cooperative async stage of chunk c into buffer buf (64 B per thread)
    auto issue_chunk = [&](int c, int buf) {
        const char* g = (const char*)xn + (size_t)c * CHUNK_BYTES + t * 64;
        unsigned    l = lds_base + (unsigned)buf * CHUNK_BYTES + (unsigned)t * 64;
        async_copy_b128(l,      g);
        async_copy_b128(l + 16, g + 16);
        async_copy_b128(l + 32, g + 32);
        async_copy_b128(l + 48, g + 48);
    };

    issue_chunk(0, 0);
    wait_async0();
    __syncthreads();

    for (int c = 0; c < NCHUNK; ++c) {
        const int buf = c & 1;
        if (c + 1 < NCHUNK) issue_chunk(c + 1, buf ^ 1);

        const unsigned short* bbase = smem + buf * (CHUNK_BYTES / 2);

        #pragma unroll
        for (int tt = 0; tt < CHUNK_TILES; ++tt) {
            const int j0 = c * CHUNK_ROWS + tt * 16;
            const unsigned short* pb = bbase + (tt * 16 + cn) * DD;

            const v16bf b0 = load_b_frag(pb,  0, lane);
            const v16bf b1 = load_b_frag(pb, 32, lane);
            const v16bf b2 = load_b_frag(pb, 64, lane);
            const v16bf b3 = load_b_frag(pb, 96, lane);

            v8f cacc = {};
            cacc = __builtin_amdgcn_wmma_f32_16x16x32_bf16(false, a0, false, b0,
                                                           (short)0, cacc, false, false);
            cacc = __builtin_amdgcn_wmma_f32_16x16x32_bf16(false, a1, false, b1,
                                                           (short)0, cacc, false, false);
            cacc = __builtin_amdgcn_wmma_f32_16x16x32_bf16(false, a2, false, b2,
                                                           (short)0, cacc, false, false);
            cacc = __builtin_amdgcn_wmma_f32_16x16x32_bf16(false, a3, false, b3,
                                                           (short)0, cacc, false, false);

            // C layout: VGPR r, lanes 0-15 -> M=r ; lanes 16-31 -> M=r+8
            float* po = out + (size_t)(i0 + mh) * NN + j0 + cn;
            #pragma unroll
            for (int r = 0; r < 8; ++r)
                po[(size_t)r * NN] = cacc[r];
        }

        // own async loads (chunk c+1) done, then block-wide: everyone finished
        // reading buf before it is overwritten in the next iteration.
        wait_async0();
        __syncthreads();
    }
}

// ---------------------------------------------------------------------------
// Kernel 3: exact per-row top-32 via 4-round radix select on order-preserving
// uint keys held in LDS (48 KB/row), then in-place masking of the row.
// One 256-thread block per row.
// ---------------------------------------------------------------------------
__global__ __launch_bounds__(256)
void knn_topk_kernel(float* __restrict__ out) {
    __shared__ unsigned keys[NN];      // 48 KB
    __shared__ unsigned hist[256];
    __shared__ unsigned sel[2];        // [0] = chosen digit, [1] = remaining rank
    __shared__ int      cnt[2];        // [0] = strictly-greater count, [1] = tie budget

    const int row = blockIdx.x;
    const int t   = threadIdx.x;
    float* rowp = out + (size_t)row * NN;

    for (int j = t; j < NN; j += 256) {
        unsigned u = __float_as_uint(rowp[j]);
        keys[j] = (u & 0x80000000u) ? ~u : (u | 0x80000000u);
    }
    __syncthreads();

    unsigned prefix = 0, pmask = 0;
    unsigned remaining = KK;

    for (int shift = 24; shift >= 0; shift -= 8) {
        hist[t & 255] = 0;
        __syncthreads();
        for (int j = t; j < NN; j += 256) {
            const unsigned key = keys[j];
            if ((key & pmask) == prefix)
                atomicAdd(&hist[(key >> shift) & 255u], 1u);
        }
        __syncthreads();
        if (t == 0) {
            unsigned acc = 0;
            int b = 255;
            for (; b > 0; --b) {
                acc += hist[b];
                if (acc >= remaining) break;
                if (b == 1) { acc += hist[0]; b = 0; break; }
            }
            sel[0] = (unsigned)b;
            sel[1] = remaining - (acc - hist[b]);
        }
        __syncthreads();
        prefix |= (sel[0] << shift);
        pmask  |= (255u << shift);
        remaining = sel[1];
        __syncthreads();
    }

    const unsigned kth = prefix;       // exact 32nd-largest key

    if (t == 0) { cnt[0] = 0; }
    __syncthreads();
    for (int j = t; j < NN; j += 256)
        if (keys[j] > kth) atomicAdd(&cnt[0], 1);
    __syncthreads();
    if (t == 0) cnt[1] = KK - cnt[0];  // tie budget at the threshold
    __syncthreads();

    for (int j = t; j < NN; j += 256) {
        const unsigned key = keys[j];
        bool keep = key > kth;
        if (!keep && key == kth) {
            keep = (atomicSub(&cnt[1], 1) > 0);
        }
        if (!keep) rowp[j] = 0.0f;
    }
}

// ---------------------------------------------------------------------------
extern "C" void kernel_launch(void* const* d_in, const int* in_sizes, int n_in,
                              void* d_out, int out_size, void* d_ws, size_t ws_size,
                              hipStream_t stream) {
    (void)in_sizes; (void)n_in; (void)out_size; (void)ws_size;

    const float* x  = (const float*)d_in[0];
    float* out      = (float*)d_out;
    unsigned short* xn = (unsigned short*)d_ws;   // NN*DD bf16 = 3 MB

    // 1) normalize rows -> bf16
    knn_normalize_kernel<<<NN, 128, 0, stream>>>(x, xn);

    // 2) full cosine similarity matrix via WMMA bf16 with async LDS staging
    //    96 blocks * 8 waves * 16 rows = 12288 rows; 32 KB dynamic LDS
    knn_gemm_kernel<<<NN / (16 * 8), 256, 2 * CHUNK_BYTES, stream>>>(xn, out);

    // 3) exact top-32 mask per row (in place)
    knn_topk_kernel<<<NN, 256, 0, stream>>>(out);
}
